// STABR_tag_encoder_26697516712561
// MI455X (gfx1250) — matrix operand order repaired and checked by reading, not compiled, not run on hardware
//
#include <hip/hip_runtime.h>
#include <hip/hip_bf16.h>

typedef __attribute__((ext_vector_type(8)))  _Float16 v8h;
typedef __attribute__((ext_vector_type(16))) _Float16 v16h;
typedef __attribute__((ext_vector_type(8)))  float    v8f;

#define B_   512
#define S_   200
#define T_   10
#define H_   64
#define NH_  192   // 3*H
#define KP_  96    // 64 (h) + 32 (x padded from 25)

__device__ __forceinline__ v8f wmma32(v16h a, v16h b, v8f c) {
  return __builtin_amdgcn_wmma_f32_16x16x32_f16(false, a, false, b, (short)0, c, false, false);
}

__device__ __forceinline__ float fast_sigmoid(float x) {
  return __builtin_amdgcn_rcpf(1.f + __expf(-x));
}

__device__ __forceinline__ float fast_tanh(float x) {
#if __has_builtin(__builtin_amdgcn_tanhf)
  return __builtin_amdgcn_tanhf(x);       // gfx1250 v_tanh_f32
#else
  float a = fabsf(x);
  float t = __expf(-2.f * a);
  float r = (1.f - t) * __builtin_amdgcn_rcpf(1.f + t);
  return copysignf(r, x);
#endif
}

// A fragment 16x32 f16: lane<16 -> row=lane, K = [k0..k0+7] and [k0+16..k0+23]
//                       lane>=16 -> row=lane-16, K = [k0+8..k0+15] and [k0+24..k0+31]
__device__ __forceinline__ v16h frag_A(const _Float16* base, int kstride, int k0, int lane) {
  int row = lane & 15;
  int kh  = (lane >> 4) << 3;
  const _Float16* p = base + row * kstride + k0 + kh;
  v8h lo = *(const v8h*)p;
  v8h hi = *(const v8h*)(p + 16);
  v16h r;
#pragma unroll
  for (int i = 0; i < 8; ++i) { r[i] = lo[i]; r[i + 8] = hi[i]; }
  return r;
}

// B fragment 32x16 f16 from an [N][K] transposed buffer:
// lane<16 -> col=lane, K = [k0..k0+7] and [k0+16..k0+23]; lane>=16 -> col=lane-16, K = [k0+8..k0+15] and [k0+24..k0+31]
__device__ __forceinline__ v16h frag_B(const _Float16* bt, int kstride, int n0, int k0, int lane) {
  int col = lane & 15;
  int kh  = (lane >> 4) << 3;
  const _Float16* p = bt + (size_t)(n0 + col) * kstride + k0 + kh;
  v8h lo = *(const v8h*)p;
  v8h hi = *(const v8h*)(p + 16);
  v16h r;
#pragma unroll
  for (int i = 0; i < 8; ++i) { r[i] = lo[i]; r[i + 8] = hi[i]; }
  return r;
}

// ---------------- Phase 1: embedding-bag masked mean -> X (B*S, 32) f16 + step mask ----------------
__global__ __launch_bounds__(128)
void embed_mean_kernel(const int* __restrict__ ids, const float* __restrict__ emb,
                       _Float16* __restrict__ X, float* __restrict__ Mb) {
  int tok  = blockIdx.x * 4 + (threadIdx.x >> 5);
  int lane = threadIdx.x & 31;
  const int* bag = ids + (size_t)tok * T_;
  float acc = 0.f, cnt = 0.f;
#pragma unroll
  for (int t = 0; t < T_; ++t) {
    int id = bag[t];                          // wave-uniform
    if (id != 0) {
      cnt += 1.f;
      if (lane < 25) acc += emb[(size_t)id * 25 + lane];
    }
  }
  float x = acc * __builtin_amdgcn_rcpf(fmaxf(cnt, 1.f));
  X[(size_t)tok * 32 + lane] = (lane < 25) ? (_Float16)x : (_Float16)0.f;
  if (lane == 0) Mb[tok] = (bag[0] != 0) ? 1.f : 0.f;
}

// ---------------- Phase 2: bidirectional GRU scan, WMMA per step ----------------
// grid.x = B/64 (4 waves x 16 rows), grid.y = 2 (dir 0 = fwd, 1 = bwd), block = 128
__global__ __launch_bounds__(128)
void gru_scan_kernel(const _Float16* __restrict__ X, const float* __restrict__ Mb,
                     const float* __restrict__ Uf, const float* __restrict__ Wf, const float* __restrict__ bf,
                     const float* __restrict__ Ub, const float* __restrict__ Wb, const float* __restrict__ bb,
                     _Float16* __restrict__ OUTh, float* __restrict__ HT) {
  const int dir = blockIdx.y;
  const float* U    = dir ? Ub : Uf;
  const float* W    = dir ? Wb : Wf;
  const float* bias = dir ? bb : bf;

  __shared__ _Float16 Ut[NH_ * KP_];          // [n][k] : k<64 = U, k 64..88 = W (padded), f16
  __shared__ float    b0s[NH_], b1s[NH_];
  __shared__ _Float16 hbuf[4][16 * 64];       // per-wave hidden state (f16 A source)
  __shared__ _Float16 xbuf[4][16 * 32];       // per-wave x staging
  __shared__ float    mbuf[4][16];

  const int tid = threadIdx.x, w = tid >> 5, lane = tid & 31;

  for (int idx = tid; idx < NH_ * KP_; idx += 128) {
    int n = idx / KP_, k = idx % KP_;
    float v = (k < 64) ? U[(size_t)k * NH_ + n]
                       : ((k - 64) < 25 ? W[(size_t)(k - 64) * NH_ + n] : 0.f);
    Ut[idx] = (_Float16)v;
  }
  for (int i = tid; i < NH_; i += 128) { b0s[i] = bias[i]; b1s[i] = bias[NH_ + i]; }
  for (int i = lane; i < 16 * 64; i += 32) hbuf[w][i] = (_Float16)0.f;
  __syncthreads();

  const int b0     = blockIdx.x * 64 + w * 16;
  const int nl     = lane & 15;
  const int half8  = (lane >> 4) << 3;
  const int dirOff = dir ? 64 : 0;
  const int s0     = dir ? (S_ - 1) : 0;
  const int sd     = dir ? -1 : 1;

  float bz0[4], bz1[4], br0[4], br1[4], bh0[4], bh1[4];
#pragma unroll
  for (int t = 0; t < 4; ++t) {
    int n = t * 16 + nl;
    bz0[t] = b0s[n];        bz1[t] = b1s[n];
    br0[t] = b0s[64 + n];   br1[t] = b1s[64 + n];
    bh0[t] = b0s[128 + n];  bh1[t] = b1s[128 + n];
  }

  float hreg[4][8];
#pragma unroll
  for (int t = 0; t < 4; ++t)
#pragma unroll
    for (int j = 0; j < 8; ++j) hreg[t][j] = 0.f;

  _Float16* hb = &hbuf[w][0];
  _Float16* xb = &xbuf[w][0];

  // Strength-reduced per-lane offsets, advanced by +/- stride each step.
  const int xrow  = lane & 15;
  const int xoffh = (lane >> 4) * 16;
  size_t xoff = ((size_t)(b0 + xrow) * S_ + s0) * 32 + xoffh;
  size_t moff = (size_t)(b0 + (lane & 15)) * S_ + s0;      // used by lanes < 16
  size_t obase[8];
#pragma unroll
  for (int j = 0; j < 8; ++j)
    obase[j] = ((size_t)(b0 + j + half8) * S_ + s0) * 128 + dirOff + nl;

  for (int step = 0; step < S_; ++step) {
    // stage x tile (16 rows x 32 halves) + step mask into per-wave LDS
    {
      const _Float16* src = X + xoff;
      *(v8h*)(xb + xrow * 32 + xoffh)     = *(const v8h*)src;
      *(v8h*)(xb + xrow * 32 + xoffh + 8) = *(const v8h*)(src + 8);
      if (lane < 16) mbuf[w][lane] = Mb[moff];
      if (step + 1 < S_) __builtin_prefetch(X + xoff + sd * 32, 0, 1);
    }
    float mrow[8];
#pragma unroll
    for (int j = 0; j < 8; ++j) mrow[j] = mbuf[w][j + half8];

    v16h a0 = frag_A(hb, 64, 0,  lane);       // h, K 0..31
    v16h a1 = frag_A(hb, 64, 32, lane);       // h, K 32..63
    v16h ax = frag_A(xb, 32, 0,  lane);       // x (padded to 32)

#pragma unroll
    for (int t = 0; t < 4; ++t) {
      v8f cz = {}; v8f cr = {}; v8f chU = {}; v8f chX = {};
      const int nz = t * 16, nr = 64 + t * 16, nh = 128 + t * 16;
      cz  = wmma32(a0, frag_B(Ut, KP_, nz, 0,  lane), cz);
      cz  = wmma32(a1, frag_B(Ut, KP_, nz, 32, lane), cz);
      cz  = wmma32(ax, frag_B(Ut, KP_, nz, 64, lane), cz);
      cr  = wmma32(a0, frag_B(Ut, KP_, nr, 0,  lane), cr);
      cr  = wmma32(a1, frag_B(Ut, KP_, nr, 32, lane), cr);
      cr  = wmma32(ax, frag_B(Ut, KP_, nr, 64, lane), cr);
      chU = wmma32(a0, frag_B(Ut, KP_, nh, 0,  lane), chU);
      chU = wmma32(a1, frag_B(Ut, KP_, nh, 32, lane), chU);
      chX = wmma32(ax, frag_B(Ut, KP_, nh, 64, lane), chX);

#pragma unroll
      for (int j = 0; j < 8; ++j) {
        float ho = hreg[t][j];
        float z  = fast_sigmoid(cz[j] + bz0[t] + bz1[t]);
        float r  = fast_sigmoid(cr[j] + br0[t] + br1[t]);
        float hh = fast_tanh(chX[j] + bh0[t] + r * (chU[j] + bh1[t]));
        float hn = z * ho + (1.f - z) * hh;
        float mk = mrow[j];
        hn = mk * hn + (1.f - mk) * ho;
        hreg[t][j] = hn;
        _Float16 hn16 = (_Float16)hn;
        hb[(j + half8) * 64 + t * 16 + nl] = hn16;
        OUTh[obase[j] + t * 16] = hn16;
      }
    }

    xoff += (size_t)(sd * 32);
    moff += (size_t)sd;
#pragma unroll
    for (int j = 0; j < 8; ++j) obase[j] += (size_t)(sd * 128);
  }

  if (dir == 0) {
#pragma unroll
    for (int t = 0; t < 4; ++t)
#pragma unroll
      for (int j = 0; j < 8; ++j)
        HT[(size_t)(b0 + j + half8) * 64 + t * 16 + nl] = hreg[t][j];
  }
}

// ---------------- Phase 3a: queries = hT_f @ W_q + b_q ----------------
__global__ __launch_bounds__(64)
void queries_kernel(const float* __restrict__ HT, const float* __restrict__ Wq,
                    const float* __restrict__ bq, float* __restrict__ Q) {
  int b = blockIdx.x, n = threadIdx.x;
  float acc = bq[n];
#pragma unroll 8
  for (int k = 0; k < 64; ++k) acc += HT[(size_t)b * 64 + k] * Wq[k * 64 + n];
  Q[(size_t)b * 64 + n] = acc;
}

// ---------------- Phase 3b: e = tanh(out @ W_k + b_k + q) @ W_e + b_e, masked ----------------
// M = B*S flattened in 16-row tiles; one wave per tile; WMMA K=128, N=64.
__global__ __launch_bounds__(128)
void scores_kernel(const _Float16* __restrict__ OUTh, const float* __restrict__ Q,
                   const float* __restrict__ Wk, const float* __restrict__ bk,
                   const float* __restrict__ We, const float* __restrict__ be,
                   const float* __restrict__ Mb, float* __restrict__ E) {
  __shared__ _Float16 WkT[64 * 128];          // [n][k] f16
  __shared__ float bkL[64], weL[64];
  __shared__ float beL;
  const int tid = threadIdx.x, w = tid >> 5, lane = tid & 31;

  for (int idx = tid; idx < 64 * 128; idx += 128) {
    int n = idx / 128, k = idx % 128;
    WkT[idx] = (_Float16)Wk[(size_t)k * 64 + n];
  }
  for (int i = tid; i < 64; i += 128) { bkL[i] = bk[i]; weL[i] = We[i]; }
  if (tid == 0) beL = be[0];
  __syncthreads();

  const int m0 = (blockIdx.x * 4 + w) * 16;
  const int nl = lane & 15;
  const int half8 = (lane >> 4) << 3;

  v8f c0 = {}; v8f c1 = {}; v8f c2 = {}; v8f c3 = {};
#pragma unroll
  for (int kk = 0; kk < 4; ++kk) {
    v16h a = frag_A(OUTh + (size_t)m0 * 128, 128, kk * 32, lane);
    c0 = wmma32(a, frag_B(WkT, 128, 0,  kk * 32, lane), c0);
    c1 = wmma32(a, frag_B(WkT, 128, 16, kk * 32, lane), c1);
    c2 = wmma32(a, frag_B(WkT, 128, 32, kk * 32, lane), c2);
    c3 = wmma32(a, frag_B(WkT, 128, 48, kk * 32, lane), c3);
  }

  float acc[8];
#pragma unroll
  for (int j = 0; j < 8; ++j) acc[j] = 0.f;
  v8f ct[4] = {c0, c1, c2, c3};
#pragma unroll
  for (int t = 0; t < 4; ++t) {
    float bk_ = bkL[t * 16 + nl], we_ = weL[t * 16 + nl];
#pragma unroll
    for (int j = 0; j < 8; ++j) {
      int m = m0 + j + half8;
      float q = Q[(size_t)(m / S_) * 64 + t * 16 + nl];
      acc[j] += fast_tanh(ct[t][j] + bk_ + q) * we_;
    }
  }
#pragma unroll
  for (int j = 0; j < 8; ++j) {
    float v = acc[j];
    v += __shfl_xor(v, 1);
    v += __shfl_xor(v, 2);
    v += __shfl_xor(v, 4);
    v += __shfl_xor(v, 8);
    if (nl == 0) {
      int m = m0 + j + half8;
      E[m] = v + beL + (1.f - Mb[m]) * -1.0e9f;
    }
  }
}

// ---------------- Phase 3c: masked softmax over S + weighted sum -> context ----------------
__global__ __launch_bounds__(128)
void attn_context_kernel(const float* __restrict__ E, const _Float16* __restrict__ OUTh,
                         float* __restrict__ out) {
  __shared__ float ew[S_];
  __shared__ float red[128];
  const int b = blockIdx.x, tid = threadIdx.x;

  float lmax = -3.4e38f;
  for (int s = tid; s < S_; s += 128) { float v = E[(size_t)b * S_ + s]; ew[s] = v; lmax = fmaxf(lmax, v); }
  red[tid] = lmax; __syncthreads();
  for (int off = 64; off > 0; off >>= 1) { if (tid < off) red[tid] = fmaxf(red[tid], red[tid + off]); __syncthreads(); }
  float mx = red[0]; __syncthreads();

  float lsum = 0.f;
  for (int s = tid; s < S_; s += 128) { float v = __expf(ew[s] - mx); ew[s] = v; lsum += v; }
  red[tid] = lsum; __syncthreads();
  for (int off = 64; off > 0; off >>= 1) { if (tid < off) red[tid] += red[tid + off]; __syncthreads(); }
  float winv = __builtin_amdgcn_rcpf(red[0]);
  __syncthreads();

  float acc = 0.f;
  const _Float16* base = OUTh + (size_t)b * S_ * 128 + tid;
  for (int s = 0; s < S_; ++s) acc += ew[s] * (float)base[(size_t)s * 128];
  out[(size_t)b * 128 + tid] = acc * winv;
}

extern "C" void kernel_launch(void* const* d_in, const int* in_sizes, int n_in,
                              void* d_out, int out_size, void* d_ws, size_t ws_size,
                              hipStream_t stream) {
  (void)in_sizes; (void)n_in; (void)out_size; (void)ws_size;
  const int*   ids = (const int*)d_in[0];
  const float* emb = (const float*)d_in[1];
  const float* Wf  = (const float*)d_in[2];
  const float* Uf  = (const float*)d_in[3];
  const float* bf  = (const float*)d_in[4];
  const float* Wb  = (const float*)d_in[5];
  const float* Ub  = (const float*)d_in[6];
  const float* bb  = (const float*)d_in[7];
  const float* Wk  = (const float*)d_in[8];
  const float* bk  = (const float*)d_in[9];
  const float* Wq  = (const float*)d_in[10];
  const float* bq  = (const float*)d_in[11];
  const float* We  = (const float*)d_in[12];
  const float* be  = (const float*)d_in[13];
  float* out = (float*)d_out;

  char* ws = (char*)d_ws;
  size_t off = 0;
  auto alignup = [](size_t x) { return (x + 255) & ~(size_t)255; };
  _Float16* X    = (_Float16*)(ws + off); off = alignup(off + (size_t)B_ * S_ * 32  * sizeof(_Float16));
  _Float16* OUTh = (_Float16*)(ws + off); off = alignup(off + (size_t)B_ * S_ * 128 * sizeof(_Float16));
  float*    Mb   = (float*)   (ws + off); off = alignup(off + (size_t)B_ * S_ * sizeof(float));
  float*    HT   = (float*)   (ws + off); off = alignup(off + (size_t)B_ * 64 * sizeof(float));
  float*    Q    = (float*)   (ws + off); off = alignup(off + (size_t)B_ * 64 * sizeof(float));
  float*    E    = (float*)   (ws + off); off = alignup(off + (size_t)B_ * S_ * sizeof(float));

  embed_mean_kernel<<<(B_ * S_) / 4, 128, 0, stream>>>(ids, emb, X, Mb);
  dim3 ggrid(B_ / 64, 2);
  gru_scan_kernel<<<ggrid, 128, 0, stream>>>(X, Mb, Uf, Wf, bf, Ub, Wb, bb, OUTh, HT);
  queries_kernel<<<B_, 64, 0, stream>>>(HT, Wq, bq, Q);
  scores_kernel<<<(B_ * S_) / 64, 128, 0, stream>>>(OUTh, Q, Wk, bk, We, be, Mb, E);
  attn_context_kernel<<<B_, 128, 0, stream>>>(E, OUTh, out);
}
